// MoE_274877907303
// MI455X (gfx1250) — matrix operand order repaired and checked by reading, not compiled, not run on hardware
//
#include <hip/hip_runtime.h>
#include <hip/hip_bf16.h>
#include <math.h>

// Problem sizes (fixed by the reference)
#define Bb   2
#define Tt   2048
#define Dd   1024
#define Hh   2048
#define Ee   8
#define KTOP 2
#define Nn   (Bb * Tt)           // 4096 tokens

typedef __attribute__((ext_vector_type(16))) __bf16 v16bf;
typedef __attribute__((ext_vector_type(8)))  __bf16 v8bf;
typedef __attribute__((ext_vector_type(8)))  float  v8f;
typedef __attribute__((ext_vector_type(4)))  int    v4i;

// ------------------------------------------------ gfx1250 async-to-LDS path
#if __has_builtin(__builtin_amdgcn_global_load_async_to_lds_b128)
#define USE_ASYNC_LDS 1
#else
#define USE_ASYNC_LDS 0
#endif

__device__ __forceinline__ void wait_async0() {
#if __has_builtin(__builtin_amdgcn_s_wait_asynccnt)
    __builtin_amdgcn_s_wait_asynccnt(0);
#else
    asm volatile("s_wait_asynccnt 0x0" ::: "memory");
#endif
}

#if USE_ASYNC_LDS
typedef __attribute__((address_space(1))) v4i* gv4p;   // global int4*
typedef __attribute__((address_space(3))) v4i* lv4p;   // LDS int4*
__device__ __forceinline__ void async_copy_b128(const void* g, void* l) {
    __builtin_amdgcn_global_load_async_to_lds_b128((gv4p)g, (lv4p)l, 0, 0);
}
#endif

// ---------------------------------------------------------------- init
__global__ void moe_init_kernel(int* __restrict__ counts) {
    if (threadIdx.x < Ee) counts[threadIdx.x] = 0;
}

// ---------------------------------------------------------------- router
__global__ void moe_router_kernel(const float* __restrict__ x,
                                  const float* __restrict__ gw,
                                  int* __restrict__ counts,
                                  int* __restrict__ tok,
                                  int* __restrict__ info_e,
                                  int* __restrict__ info_s,
                                  float* __restrict__ info_w) {
    const int n = blockIdx.x * blockDim.x + threadIdx.x;
    if (n >= Nn) return;
    const float4* xr = (const float4*)(x + (size_t)n * Dd);
    float l[Ee];
#pragma unroll
    for (int e = 0; e < Ee; ++e) {
        const float4* g = (const float4*)(gw + (size_t)e * Dd);
        float acc = 0.f;
        for (int i = 0; i < Dd / 4; ++i) {
            float4 a = xr[i], b = g[i];
            acc += a.x * b.x + a.y * b.y + a.z * b.z + a.w * b.w;
        }
        l[e] = acc;
    }
    float m = l[0];
#pragma unroll
    for (int e = 1; e < Ee; ++e) m = fmaxf(m, l[e]);
    int i0 = 0; float b0 = l[0];
#pragma unroll
    for (int e = 1; e < Ee; ++e) if (l[e] > b0) { b0 = l[e]; i0 = e; }
    int i1 = -1; float b1 = -INFINITY;
#pragma unroll
    for (int e = 0; e < Ee; ++e) if (e != i0 && l[e] > b1) { b1 = l[e]; i1 = e; }
    const float p0 = __expf(b0 - m), p1 = __expf(b1 - m);
    const float inv = 1.f / (p0 + p1);
    const int s0 = atomicAdd(&counts[i0], 1);
    const int s1 = atomicAdd(&counts[i1], 1);
    tok[i0 * Nn + s0] = n;
    tok[i1 * Nn + s1] = n;
    info_e[n * 2 + 0] = i0; info_s[n * 2 + 0] = s0; info_w[n * 2 + 0] = p0 * inv;
    info_e[n * 2 + 1] = i1; info_s[n * 2 + 1] = s1; info_w[n * 2 + 1] = p1 * inv;
}

// ---------------------------------------------------------------- prefix
__global__ void moe_prefix_kernel(const int* __restrict__ counts, int* __restrict__ offs) {
    if (threadIdx.x == 0) {
        int a = 0;
        for (int e = 0; e < Ee; ++e) { offs[e] = a; a += counts[e]; }
    }
}

// ---------------------------------------------------------------- tiling
#define TM 128
#define TN 64
#define TK 32
#define LP (TK + 8)   // LDS row pitch in bf16 (80B: 16B-aligned, conflict-skewed)

// ---------------------------------------------------------------- stage 1
// h = silu(x@w1^T) * (x@w3^T) per expert; bf16 WMMA, fp32 accumulate.
// Ping-pong LDS + register prefetch: one barrier per K-step, next K-tile's
// global loads issued before the WMMA phase so HBM latency hides under MMA.
__global__ __launch_bounds__(256)
void moe_gemm1_kernel(const float* __restrict__ x,
                      const float* __restrict__ w1,
                      const float* __restrict__ w3,
                      const int* __restrict__ counts,
                      const int* __restrict__ offs,
                      const int* __restrict__ tok,
                      __bf16* __restrict__ hbuf) {
    const int e   = blockIdx.z;
    const int cnt = counts[e];
    const int m0  = blockIdx.x * TM;
    if (m0 >= cnt) return;
    const int h0  = blockIdx.y * TN;

    __shared__ __bf16 As[2][TM][LP];     // [buf][row][k]
    __shared__ __bf16 B1s[2][TN][LP];    // [buf][col][k]  (col = h index)
    __shared__ __bf16 B3s[2][TN][LP];

    const int t    = threadIdx.x;
    const int lane = t & 31;
    const int wave = t >> 5;
    const int wm   = wave & 3;
    const int wn   = wave >> 2;
    const int half = lane >> 4;
    const int lr   = lane & 15;

    // load assignments (constant over K loop)
    const int arow  = t >> 1;
    const int apart = t & 1;
    const int atok  = (m0 + arow < cnt) ? tok[e * Nn + m0 + arow] : -1;
    const int bmat  = t >> 7;
    const int brow  = (t >> 1) & 63;
    const int bpart = t & 1;
    const float* wbase = (bmat ? w3 : w1) + (((size_t)e * Hh + (h0 + brow)) * Dd);

    float4 ra[4], rb[4];
#pragma unroll
    for (int q = 0; q < 4; ++q) { ra[q] = make_float4(0.f, 0.f, 0.f, 0.f); }

    auto loadA = [&](int k0) {
        if (atok >= 0) {
            const float4* src = (const float4*)(x + (size_t)atok * Dd + k0 + apart * 16);
#pragma unroll
            for (int q = 0; q < 4; ++q) ra[q] = src[q];
        }
    };
    auto loadB = [&](int k0) {
        const float4* src = (const float4*)(wbase + k0 + bpart * 16);
#pragma unroll
        for (int q = 0; q < 4; ++q) rb[q] = src[q];
    };

    v8f acc1[2][2], acc3[2][2];
#pragma unroll
    for (int i = 0; i < 2; ++i)
#pragma unroll
        for (int j = 0; j < 2; ++j) {
            v8f z = {0.f, 0.f, 0.f, 0.f, 0.f, 0.f, 0.f, 0.f};
            acc1[i][j] = z; acc3[i][j] = z;
        }

    loadA(0); loadB(0);

    const int NS = Dd / TK;
    int buf = 0;
    for (int s = 0; s < NS; ++s) {
        // store prefetched regs into LDS[buf] (fp32 -> bf16 here, once per elem)
        {
            __bf16* dst = &As[buf][arow][apart * 16];
#pragma unroll
            for (int q = 0; q < 4; ++q) {
                float4 v = ra[q];
                dst[q * 4 + 0] = (__bf16)v.x; dst[q * 4 + 1] = (__bf16)v.y;
                dst[q * 4 + 2] = (__bf16)v.z; dst[q * 4 + 3] = (__bf16)v.w;
            }
        }
        {
            __bf16* dst = bmat ? &B3s[buf][brow][bpart * 16] : &B1s[buf][brow][bpart * 16];
#pragma unroll
            for (int q = 0; q < 4; ++q) {
                float4 v = rb[q];
                dst[q * 4 + 0] = (__bf16)v.x; dst[q * 4 + 1] = (__bf16)v.y;
                dst[q * 4 + 2] = (__bf16)v.z; dst[q * 4 + 3] = (__bf16)v.w;
            }
        }
        __syncthreads();

        // prefetch next K-tile (latency hidden by the WMMA phase below)
        if (s + 1 < NS) { loadA((s + 1) * TK); loadB((s + 1) * TK); }

#pragma unroll
        for (int fm = 0; fm < 2; ++fm) {
            union { v16bf v; v8bf h[2]; } a;
            const int r = wm * 32 + fm * 16 + lr;
            a.h[0] = *(const v8bf*)&As[buf][r][half * 8];
            a.h[1] = *(const v8bf*)&As[buf][r][16 + half * 8];
#pragma unroll
            for (int fn = 0; fn < 2; ++fn) {
                const int c = wn * 32 + fn * 16 + lr;
                union { v16bf v; v8bf h[2]; } b1, b3;
                b1.h[0] = *(const v8bf*)&B1s[buf][c][half * 8];
                b1.h[1] = *(const v8bf*)&B1s[buf][c][16 + half * 8];
                b3.h[0] = *(const v8bf*)&B3s[buf][c][half * 8];
                b3.h[1] = *(const v8bf*)&B3s[buf][c][16 + half * 8];
                acc1[fm][fn] = __builtin_amdgcn_wmma_f32_16x16x32_bf16(
                    false, a.v, false, b1.v, (short)0, acc1[fm][fn], false, false);
                acc3[fm][fn] = __builtin_amdgcn_wmma_f32_16x16x32_bf16(
                    false, a.v, false, b3.v, (short)0, acc3[fm][fn], false, false);
            }
        }
        buf ^= 1;
    }

    // fused SiLU(h1) * h3 -> bf16 scratch (row = expert slot)
    const int ob = offs[e];
#pragma unroll
    for (int fm = 0; fm < 2; ++fm)
#pragma unroll
        for (int fn = 0; fn < 2; ++fn) {
            v8f h1v = acc1[fm][fn], h3v = acc3[fm][fn];
            const int col = h0 + wn * 32 + fn * 16 + lr;
#pragma unroll
            for (int i = 0; i < 8; ++i) {
                const int row = m0 + wm * 32 + fm * 16 + half * 8 + i; // M = i + 8*half
                if (row < cnt) {
                    float h1 = h1v[i], h3 = h3v[i];
                    float g = h1 / (1.f + __expf(-h1)) * h3;
                    hbuf[(size_t)(ob + row) * Hh + col] = (__bf16)g;
                }
            }
        }
}

// ---------------------------------------------------------------- stage 2
// y = h @ w2^T per expert. A tile is already bf16 -> moved with
// GLOBAL_LOAD_ASYNC_TO_LDS_B128 (ASYNCcnt) into the ping-pong buffer while
// WMMAs run; B tile (fp32 w2) via register prefetch + cvt.
__global__ __launch_bounds__(256)
void moe_gemm2_kernel(const __bf16* __restrict__ hbuf,
                      const float* __restrict__ w2,
                      const int* __restrict__ counts,
                      const int* __restrict__ offs,
                      float* __restrict__ ybuf) {
    const int e   = blockIdx.z;
    const int cnt = counts[e];
    const int m0  = blockIdx.x * TM;
    if (m0 >= cnt) return;
    const int d0  = blockIdx.y * TN;
    const int ob  = offs[e];

    __shared__ __bf16 As[2][TM][LP];
    __shared__ __bf16 Bs[2][TN][LP];

    const int t    = threadIdx.x;
    const int lane = t & 31;
    const int wave = t >> 5;
    const int wm   = wave & 3;
    const int wn   = wave >> 2;
    const int half = lane >> 4;
    const int lr   = lane & 15;

    // A copy assignment: 2 threads per row, 32B (2 x b128 chunks) each
    const int  crow  = t >> 1;
    const int  cpart = t & 1;
    const __bf16* asrc = hbuf + (size_t)(ob + m0 + crow) * Hh + cpart * 16;
    // B load assignment: threads 128..255, 2 per w2 row
    const bool doB   = (t >= 128);
    const int  brow  = (t >> 1) & 63;
    const int  bpart = t & 1;
    const float* wbase = w2 + (((size_t)e * Dd + (d0 + brow)) * Hh);

    float4 rb[4];
#if !USE_ASYNC_LDS
    uint4 raw[2];
    const bool arowok = (m0 + crow < cnt);
#endif

    auto loadB = [&](int k0) {
        if (doB) {
            const float4* src = (const float4*)(wbase + k0 + bpart * 16);
#pragma unroll
            for (int q = 0; q < 4; ++q) rb[q] = src[q];
        }
    };

#if USE_ASYNC_LDS
    auto issueA = [&](int b, int k0) {
        // rows >= cnt read harmless neighboring workspace; never stored back
        async_copy_b128((const void*)(asrc + k0),
                        (void*)&As[b][crow][cpart * 16]);
        async_copy_b128((const void*)(asrc + k0 + 8),
                        (void*)&As[b][crow][cpart * 16 + 8]);
    };
#else
    auto loadA = [&](int k0) {
        if (arowok) {
            const uint4* src = (const uint4*)(asrc + k0);
            raw[0] = src[0]; raw[1] = src[1];
        } else {
            uint4 z = {0u, 0u, 0u, 0u};
            raw[0] = z; raw[1] = z;
        }
    };
#endif

    v8f acc[2][2];
#pragma unroll
    for (int i = 0; i < 2; ++i)
#pragma unroll
        for (int j = 0; j < 2; ++j) {
            v8f z = {0.f, 0.f, 0.f, 0.f, 0.f, 0.f, 0.f, 0.f};
            acc[i][j] = z;
        }

#if USE_ASYNC_LDS
    issueA(0, 0);
#else
    loadA(0);
#endif
    loadB(0);

    const int NS = Hh / TK;
    int buf = 0;
    for (int s = 0; s < NS; ++s) {
#if USE_ASYNC_LDS
        wait_async0();                       // A tile for this step landed in LDS[buf]
#else
        {
            uint4* dst = (uint4*)&As[buf][crow][cpart * 16];
            dst[0] = raw[0]; dst[1] = raw[1];
        }
#endif
        if (doB) {
            __bf16* dst = &Bs[buf][brow][bpart * 16];
#pragma unroll
            for (int q = 0; q < 4; ++q) {
                float4 v = rb[q];
                dst[q * 4 + 0] = (__bf16)v.x; dst[q * 4 + 1] = (__bf16)v.y;
                dst[q * 4 + 2] = (__bf16)v.z; dst[q * 4 + 3] = (__bf16)v.w;
            }
        }
        __syncthreads();

        if (s + 1 < NS) {
#if USE_ASYNC_LDS
            issueA(buf ^ 1, (s + 1) * TK);   // overlaps with WMMAs below
#else
            loadA((s + 1) * TK);
#endif
            loadB((s + 1) * TK);
        }

#pragma unroll
        for (int fm = 0; fm < 2; ++fm) {
            union { v16bf v; v8bf h[2]; } a;
            const int r = wm * 32 + fm * 16 + lr;
            a.h[0] = *(const v8bf*)&As[buf][r][half * 8];
            a.h[1] = *(const v8bf*)&As[buf][r][16 + half * 8];
#pragma unroll
            for (int fn = 0; fn < 2; ++fn) {
                const int c = wn * 32 + fn * 16 + lr;
                union { v16bf v; v8bf h[2]; } b;
                b.h[0] = *(const v8bf*)&Bs[buf][c][half * 8];
                b.h[1] = *(const v8bf*)&Bs[buf][c][16 + half * 8];
                acc[fm][fn] = __builtin_amdgcn_wmma_f32_16x16x32_bf16(
                    false, a.v, false, b.v, (short)0, acc[fm][fn], false, false);
            }
        }
        buf ^= 1;
    }

#pragma unroll
    for (int fm = 0; fm < 2; ++fm)
#pragma unroll
        for (int fn = 0; fn < 2; ++fn) {
            v8f v = acc[fm][fn];
            const int col = d0 + wn * 32 + fn * 16 + lr;
#pragma unroll
            for (int i = 0; i < 8; ++i) {
                const int row = m0 + wm * 32 + fm * 16 + half * 8 + i;
                if (row < cnt) ybuf[(size_t)(ob + row) * Dd + col] = v[i];
            }
        }
}

// ---------------------------------------------------------------- combine
__global__ void moe_combine_kernel(const float* __restrict__ ybuf,
                                   const int* __restrict__ offs,
                                   const int* __restrict__ info_e,
                                   const int* __restrict__ info_s,
                                   const float* __restrict__ info_w,
                                   float* __restrict__ out) {
    const int n  = blockIdx.x;
    const int r0 = offs[info_e[n * 2 + 0]] + info_s[n * 2 + 0];
    const int r1 = offs[info_e[n * 2 + 1]] + info_s[n * 2 + 1];
    const float w0 = info_w[n * 2 + 0], w1 = info_w[n * 2 + 1];
    const float4* y0 = (const float4*)(ybuf + (size_t)r0 * Dd);
    const float4* y1 = (const float4*)(ybuf + (size_t)r1 * Dd);
    float4* o = (float4*)(out + (size_t)n * Dd);
    for (int i = threadIdx.x; i < Dd / 4; i += blockDim.x) {
        float4 a = y0[i], b = y1[i], r;
        r.x = w0 * a.x + w1 * b.x; r.y = w0 * a.y + w1 * b.y;
        r.z = w0 * a.z + w1 * b.z; r.w = w0 * a.w + w1 * b.w;
        o[i] = r;
    }
}

// ---------------------------------------------------------------- launch
extern "C" void kernel_launch(void* const* d_in, const int* in_sizes, int n_in,
                              void* d_out, int out_size, void* d_ws, size_t ws_size,
                              hipStream_t stream) {
    const float* x  = (const float*)d_in[0];
    const float* gw = (const float*)d_in[1];
    const float* w1 = (const float*)d_in[2];
    const float* w2 = (const float*)d_in[3];
    const float* w3 = (const float*)d_in[4];
    float* out = (float*)d_out;

    // carve workspace (~67.3 MB total)
    size_t off = 0;
    auto carve = [&](size_t bytes) -> void* {
        off = (off + 255) & ~(size_t)255;
        void* p = (char*)d_ws + off;
        off += bytes;
        return p;
    };
    int*    counts = (int*)carve(Ee * sizeof(int));
    int*    eoffs  = (int*)carve(Ee * sizeof(int));
    int*    tok    = (int*)carve((size_t)Ee * Nn * sizeof(int));
    int*    info_e = (int*)carve((size_t)Nn * KTOP * sizeof(int));
    int*    info_s = (int*)carve((size_t)Nn * KTOP * sizeof(int));
    float*  info_w = (float*)carve((size_t)Nn * KTOP * sizeof(float));
    __bf16* hbuf   = (__bf16*)carve((size_t)Nn * KTOP * Hh * sizeof(__bf16)); // 32 MB
    float*  ybuf   = (float*)carve((size_t)Nn * KTOP * Dd * sizeof(float));   // 32 MB
    (void)ws_size; (void)in_sizes; (void)n_in; (void)out_size;

    moe_init_kernel<<<1, 32, 0, stream>>>(counts);
    moe_router_kernel<<<(Nn + 255) / 256, 256, 0, stream>>>(x, gw, counts, tok,
                                                            info_e, info_s, info_w);
    moe_prefix_kernel<<<1, 1, 0, stream>>>(counts, eoffs);

    dim3 g1(Nn / TM, Hh / TN, Ee);   // (32, 32, 8), most blocks early-exit
    moe_gemm1_kernel<<<g1, 256, 0, stream>>>(x, w1, w3, counts, eoffs, tok, hbuf);

    dim3 g2(Nn / TM, Dd / TN, Ee);   // (32, 16, 8)
    moe_gemm2_kernel<<<g2, 256, 0, stream>>>(hbuf, w2, counts, eoffs, ybuf);

    moe_combine_kernel<<<Nn, 256, 0, stream>>>(ybuf, eoffs, info_e, info_s, info_w, out);
}
